// AttnNet_12111807774877
// MI455X (gfx1250) — matrix operand order repaired and checked by reading, not compiled, not run on hardware
//
#include <hip/hip_runtime.h>

#define B_   16
#define S_   1024
#define EMB_ 300
#define KP_  320     // EMB padded to multiple of 32 for bf16 WMMA K-steps
#define HID_ 128
#define G4_  512     // 4*HID
#define D_   256     // 2*HID
#define D2_  512     // 2*D
#define OUT_ 128

typedef __attribute__((ext_vector_type(16))) __bf16 v16bf;
typedef __attribute__((ext_vector_type(8)))  float  v8f;
typedef __attribute__((ext_vector_type(4)))  int    v4i_;

union FragBF {
  uint4          q[2];
  unsigned short h[16];
  v16bf          v;
};

__device__ __forceinline__ unsigned short f2b(float f) {
  union { float f; unsigned u; } x; x.f = f;
  unsigned r = x.u + 0x7FFFu + ((x.u >> 16) & 1u);   // RNE truncate to bf16
  return (unsigned short)(r >> 16);
}
__device__ __forceinline__ float b2f(unsigned short h) {
  union { unsigned u; float f; } x; x.u = ((unsigned)h) << 16;
  return x.f;
}
__device__ __forceinline__ float sigmoidf_(float x) { return 1.f / (1.f + __expf(-x)); }

__device__ __forceinline__ v8f wmma_bf16(const FragBF& a, const FragBF& b, v8f c) {
  // D = A(16x32 bf16) * B(32x16 bf16) + C(16x16 f32)
  return __builtin_amdgcn_wmma_f32_16x16x32_bf16(false, a.v, false, b.v, (short)0, c,
                                                 false, false);
}

// ---- CDNA5 async global->LDS copy (ASYNCcnt-tracked), with safe fallback ----
#if __has_builtin(__builtin_amdgcn_global_load_async_to_lds_b128)
#define HAVE_ASYNC_LDS 1
#else
#define HAVE_ASYNC_LDS 0
#endif

__device__ __forceinline__ void async_copy16(void* lds_dst, const void* gsrc) {
#if HAVE_ASYNC_LDS
  __builtin_amdgcn_global_load_async_to_lds_b128((v4i_*)gsrc, (v4i_*)lds_dst, 0, 0);
#else
  *(uint4*)lds_dst = *(const uint4*)gsrc;                 // synchronous fallback
#endif
}
__device__ __forceinline__ void async_wait0() {
#if HAVE_ASYNC_LDS
#if __has_builtin(__builtin_amdgcn_s_wait_asynccnt)
  __builtin_amdgcn_s_wait_asynccnt(0);
#else
  asm volatile("s_wait_asynccnt 0x0" ::: "memory");
#endif
#endif
}

// ---------------------------------------------------------------------------
// K0: convert weights fp32 -> bf16 (row-major, w_ih padded K 300->320), zero pooled
// ---------------------------------------------------------------------------
__global__ void pack_weights(const float* wih_f, const float* whh_f,
                             const float* wih_b, const float* whh_b,
                             const float* Wq,    const float* Wo,
                             unsigned short* p_wih_f, unsigned short* p_whh_f,
                             unsigned short* p_wih_b, unsigned short* p_whh_b,
                             unsigned short* p_wq,    unsigned short* p_wo,
                             float* pooled)
{
  int tid = blockIdx.x * blockDim.x + threadIdx.x;
  int nth = gridDim.x * blockDim.x;
  for (int i = tid; i < G4_ * KP_; i += nth) {
    int r = i / KP_, c = i % KP_;
    p_wih_f[i] = (c < EMB_) ? f2b(wih_f[r * EMB_ + c]) : 0;
    p_wih_b[i] = (c < EMB_) ? f2b(wih_b[r * EMB_ + c]) : 0;
  }
  for (int i = tid; i < G4_ * HID_; i += nth) {
    p_whh_f[i] = f2b(whh_f[i]);
    p_whh_b[i] = f2b(whh_b[i]);
  }
  for (int i = tid; i < D_ * D_;  i += nth) p_wq[i] = f2b(Wq[i]);
  for (int i = tid; i < D_ * D2_; i += nth) p_wo[i] = f2b(Wo[i]);
  for (int i = tid; i < B_ * D_;  i += nth) pooled[i] = 0.f;
}

// ---------------------------------------------------------------------------
// K1: embedding gather -> bf16, padded to KP_ (one token per block)
// ---------------------------------------------------------------------------
__global__ void embed_gather(const int* __restrict__ ids,
                             const float* __restrict__ table,
                             unsigned short* __restrict__ out)
{
  int tok = blockIdx.x;                      // b*S + s
  int id  = ids[tok];
  const float* src = table + (long)id * EMB_;
  unsigned short* dst = out + (long)tok * KP_;
  for (int c = threadIdx.x; c < KP_; c += blockDim.x)
    dst[c] = (c < EMB_) ? f2b(src[c]) : (unsigned short)0;
}

// ---------------------------------------------------------------------------
// K2: xg[dir][s][b][512] = embeds[b][s][:] @ w_ih[dir]^T + b_ih + b_hh  (bf16 out)
// grid (4 ntile-groups, 1024 s, 2 dir), 256 thr = 8 waves, one 16x16 tile/wave
// ---------------------------------------------------------------------------
__global__ __launch_bounds__(256) void xg_gemm(
    const unsigned short* __restrict__ embeds,
    const unsigned short* __restrict__ wih_f, const unsigned short* __restrict__ wih_b,
    const float* __restrict__ bih_f, const float* __restrict__ bhh_f,
    const float* __restrict__ bih_b, const float* __restrict__ bhh_b,
    unsigned short* __restrict__ xg_f, unsigned short* __restrict__ xg_b)
{
  const unsigned short* wih = blockIdx.z ? wih_b : wih_f;
  const float* bi = blockIdx.z ? bih_b : bih_f;
  const float* bh = blockIdx.z ? bhh_b : bhh_f;
  unsigned short* xg = blockIdx.z ? xg_b : xg_f;

  int lane = threadIdx.x & 31, w = threadIdx.x >> 5;
  int lm = lane & 15, lh = lane >> 4;
  int nt = blockIdx.x * 8 + w;               // 0..31 -> gate column tile
  int s  = blockIdx.y;
  int n  = nt * 16 + lm;

  const unsigned short* arow = embeds + ((long)lm * S_ + s) * KP_ + 8 * lh;
  const unsigned short* brow = wih + (long)n * KP_ + 8 * lh;
  __builtin_prefetch(brow, 0, 1);

  v8f acc = {};
#pragma unroll
  for (int kk = 0; kk < KP_ / 32; ++kk) {
    FragBF a, b;
    a.q[0] = *(const uint4*)(arow + kk * 32);
    a.q[1] = *(const uint4*)(arow + kk * 32 + 16);
    b.q[0] = *(const uint4*)(brow + kk * 32);
    b.q[1] = *(const uint4*)(brow + kk * 32 + 16);
    acc = wmma_bf16(a, b, acc);
  }
  float bias = bi[n] + bh[n];
  unsigned short* xrow = xg + ((long)s * 16 + 8 * lh) * G4_ + n;  // row m = i + 8*lh
#pragma unroll
  for (int i = 0; i < 8; ++i) xrow[i * G4_] = f2b(acc[i] + bias);
}

// ---------------------------------------------------------------------------
// K3: sequential LSTM scan, one workgroup per direction (1024 thr = 32 waves).
// w_hh in registers as pre-gathered WMMA B-fragments for the entire scan.
// Next step's 16x512 xg panel is async-copied into double-buffered dynamic LDS
// (ASYNCcnt) while the current step's 4-WMMA chain + cell update run.
// ---------------------------------------------------------------------------
__global__ __launch_bounds__(1024, 1) void lstm_scan(
    const unsigned short* __restrict__ xg_f, const unsigned short* __restrict__ xg_b,
    const unsigned short* __restrict__ whh_f, const unsigned short* __restrict__ whh_b,
    unsigned short* __restrict__ ctx)
{
  __shared__ __align__(16) unsigned short h_lds[B_ * HID_];   // bf16 [16][128]
  __shared__ float c_state[B_ * HID_];                        // f32  [16][128]
  __shared__ float gates[B_ * G4_];                           // f32  [16][512]
  extern __shared__ __align__(16) char dynls[];               // 2 x 16KB xg panels
  unsigned short* xgp0 = (unsigned short*)dynls;
  unsigned short* xgp1 = (unsigned short*)(dynls + (size_t)B_ * G4_ * 2);

  int dir = blockIdx.x;
  const unsigned short* xg = dir ? xg_b : xg_f;
  const unsigned short* whh = dir ? whh_b : whh_f;

  int lane = threadIdx.x & 31, w = threadIdx.x >> 5;          // wave = gate ntile
  int lm = lane & 15, lh = lane >> 4;
  int n = w * 16 + lm;

  for (int i = threadIdx.x; i < B_ * HID_; i += blockDim.x) {
    h_lds[i] = 0; c_state[i] = 0.f;
  }

  // preload w_hh B-fragments (K=128 -> 4 ksteps), resident for all 1024 steps
  FragBF bw[4];
  {
    const unsigned short* brow = whh + (long)n * HID_ + 8 * lh;
#pragma unroll
    for (int kk = 0; kk < 4; ++kk) {
      bw[kk].q[0] = *(const uint4*)(brow + kk * 32);
      bw[kk].q[1] = *(const uint4*)(brow + kk * 32 + 16);
    }
  }

  // stage panel for t=0   (16KB = 1024 x 16B chunks, 1 per thread)
  {
    int s0 = dir ? (S_ - 1) : 0;
    const unsigned short* src = xg + (long)s0 * 16 * G4_;
    int c = threadIdx.x;
    async_copy16(xgp0 + c * 8, src + c * 8);
  }
  async_wait0();
  __syncthreads();

  for (int t = 0; t < S_; ++t) {
    int s = dir ? (S_ - 1 - t) : t;
    unsigned short* cur = (t & 1) ? xgp1 : xgp0;
    unsigned short* nxt = (t & 1) ? xgp0 : xgp1;

    // kick off async stage of next panel; it completes during this step's math
    if (t + 1 < S_) {
      int s1 = dir ? (S_ - 2 - t) : (t + 1);
      const unsigned short* src = xg + (long)s1 * 16 * G4_;
      int c = threadIdx.x;
      async_copy16(nxt + c * 8, src + c * 8);
    }

    // gates tile = h(16x128) @ w_hh^T (+ xg_t from LDS panel)
    const unsigned short* arow = h_lds + lm * HID_ + 8 * lh;
    v8f acc = {};
#pragma unroll
    for (int kk = 0; kk < 4; ++kk) {
      FragBF a;
      a.q[0] = *(const uint4*)(arow + kk * 32);
      a.q[1] = *(const uint4*)(arow + kk * 32 + 16);
      acc = wmma_bf16(a, bw[kk], acc);
    }
    const unsigned short* xrow = cur + (8 * lh) * G4_ + n;
#pragma unroll
    for (int i = 0; i < 8; ++i)
      gates[(i + 8 * lh) * G4_ + n] = acc[i] + b2f(xrow[i * G4_]);
    __syncthreads();

    // elementwise cell update: 2048 values / 1024 threads
#pragma unroll
    for (int r = 0; r < 2; ++r) {
      int idx = threadIdx.x + r * 1024;
      int m = idx >> 7, j = idx & 127;
      float ig = sigmoidf_(gates[m * G4_ + j]);
      float fg = sigmoidf_(gates[m * G4_ + 128 + j]);
      float gg = tanhf   (gates[m * G4_ + 256 + j]);
      float og = sigmoidf_(gates[m * G4_ + 384 + j]);
      float c = fg * c_state[m * HID_ + j] + ig * gg;
      float h = og * tanhf(c);
      c_state[m * HID_ + j] = c;
      h_lds[m * HID_ + j] = f2b(h);
      ctx[((long)m * S_ + s) * D_ + dir * HID_ + j] = f2b(h);
    }
    async_wait0();          // next panel resident before anyone reads it
    __syncthreads();
  }
}

// ---------------------------------------------------------------------------
// K4: ctx [b][s][d] -> ctxT [b][d][s]  (bf16, 32x32 LDS tiles)
// ---------------------------------------------------------------------------
__global__ void ctx_transpose(const unsigned short* __restrict__ ctx,
                              unsigned short* __restrict__ ctxT)
{
  __shared__ unsigned short tile[32][33];
  int s0 = blockIdx.x * 32, d0 = blockIdx.y * 32, b = blockIdx.z;
  int tx = threadIdx.x, ty = threadIdx.y;
#pragma unroll
  for (int i = 0; i < 4; ++i) {
    int sr = ty + i * 8;
    tile[sr][tx] = ctx[((long)b * S_ + s0 + sr) * D_ + d0 + tx];
  }
  __syncthreads();
#pragma unroll
  for (int i = 0; i < 4; ++i) {
    int dr = ty + i * 8;
    ctxT[((long)b * D_ + d0 + dr) * S_ + s0 + tx] = tile[tx][dr];
  }
}

// ---------------------------------------------------------------------------
// K5: fused attention + output projection + mean-pool accumulation.
// One block per (16-query tile, batch). Query ctx tile async-staged to LDS;
// scores live only in LDS (bf16), never touch HBM.
// ---------------------------------------------------------------------------
__global__ __launch_bounds__(256, 2) void attn_kernel(
    const unsigned short* __restrict__ ctx,  const unsigned short* __restrict__ ctxT,
    const unsigned short* __restrict__ Wq,   const unsigned short* __restrict__ Wo,
    const unsigned char* __restrict__ mask,  float* __restrict__ pooled)
{
  __shared__ __align__(16) unsigned short cq[16 * 256];     // query ctx tile (bf16)
  __shared__ __align__(16) unsigned short qw[16 * 256];     // q, then reused: weighted
  __shared__ __align__(16) unsigned short attnS[16 * 1024]; // scores -> probs (bf16)
  __shared__ float red[16][16];
  __shared__ float fm[16], fs[16];

  int lane = threadIdx.x & 31, w = threadIdx.x >> 5;
  int lm = lane & 15, lh = lane >> 4;
  int stile = blockIdx.x, b = blockIdx.y;
  const unsigned short* ctxb = ctx + (long)b * S_ * D_;
  const unsigned char* maskb = mask + (long)b * S_;

  // ---- async stage of the 16x256 query tile (8KB = 512 x 16B chunks) ----
  for (int c = threadIdx.x; c < 512; c += blockDim.x) {
    int row = c >> 5, chunk = c & 31;
    async_copy16(&cq[row * 256 + chunk * 8],
                 ctxb + (long)(stile * 16 + row) * D_ + chunk * 8);
  }
  async_wait0();
  __syncthreads();

  // A rows for this query tile, now in LDS (query m = lm)
  const unsigned short* arow0 = cq + (long)lm * D_ + 8 * lh;

  // ---- Phase 1: q = ctx_q @ Wq^T ; each wave owns 2 of 16 n-tiles ----
#pragma unroll
  for (int t = 0; t < 2; ++t) {
    int nt = w * 2 + t;
    const unsigned short* brow = Wq + (long)(nt * 16 + lm) * D_ + 8 * lh;
    v8f acc = {};
#pragma unroll
    for (int kk = 0; kk < 8; ++kk) {
      FragBF a, bb;
      a.q[0]  = *(const uint4*)(arow0 + kk * 32);
      a.q[1]  = *(const uint4*)(arow0 + kk * 32 + 16);
      bb.q[0] = *(const uint4*)(brow + kk * 32);
      bb.q[1] = *(const uint4*)(brow + kk * 32 + 16);
      acc = wmma_bf16(a, bb, acc);
    }
#pragma unroll
    for (int i = 0; i < 8; ++i) qw[(i + 8 * lh) * D_ + nt * 16 + lm] = f2b(acc[i]);
  }
  __syncthreads();

  // ---- Phase 2: scores = q @ ctx^T (masked), stored bf16 in LDS ----
  FragBF qa[8];
  {
    const unsigned short* qrow = qw + lm * D_ + 8 * lh;
#pragma unroll
    for (int kk = 0; kk < 8; ++kk) {
      qa[kk].q[0] = *(const uint4*)(qrow + kk * 32);
      qa[kk].q[1] = *(const uint4*)(qrow + kk * 32 + 16);
    }
  }
  for (int kt = w * 8; kt < w * 8 + 8; ++kt) {       // 8 key tiles per wave
    int key = kt * 16 + lm;
    const unsigned short* brow = ctxb + (long)key * D_ + 8 * lh;
    __builtin_prefetch(brow + D_ * 16, 0, 1);
    v8f acc = {};
#pragma unroll
    for (int kk = 0; kk < 8; ++kk) {
      FragBF bb;
      bb.q[0] = *(const uint4*)(brow + kk * 32);
      bb.q[1] = *(const uint4*)(brow + kk * 32 + 16);
      acc = wmma_bf16(qa[kk], bb, acc);
    }
    bool mk = maskb[key] != 0;
#pragma unroll
    for (int i = 0; i < 8; ++i)
      attnS[(i + 8 * lh) * S_ + key] = f2b(mk ? -1e9f : acc[i]);
  }
  __syncthreads();

  // ---- Phase 3: softmax over 1024 keys per row, in LDS ----
  {
    int m3 = threadIdx.x >> 4, c3 = threadIdx.x & 15;
    float lmax = -3e38f;
    for (int j = c3; j < S_; j += 16) lmax = fmaxf(lmax, b2f(attnS[m3 * S_ + j]));
    red[m3][c3] = lmax;
    __syncthreads();
    if (threadIdx.x < 16) {
      float mx = -3e38f;
      for (int j = 0; j < 16; ++j) mx = fmaxf(mx, red[threadIdx.x][j]);
      fm[threadIdx.x] = mx;
    }
    __syncthreads();
    float mrow = fm[m3], lsum = 0.f;
    for (int j = c3; j < S_; j += 16) lsum += __expf(b2f(attnS[m3 * S_ + j]) - mrow);
    red[m3][c3] = lsum;
    __syncthreads();
    if (threadIdx.x < 16) {
      float sm = 0.f;
      for (int j = 0; j < 16; ++j) sm += red[threadIdx.x][j];
      fs[threadIdx.x] = sm;
    }
    __syncthreads();
    float inv = 1.f / fs[m3];
    for (int j = c3; j < S_; j += 16)
      attnS[m3 * S_ + j] = f2b(__expf(b2f(attnS[m3 * S_ + j]) - mrow) * inv);
  }
  __syncthreads();

  // ---- Phase 4: weighted = attn @ ctx  (B from ctxT: contiguous k) ----
#pragma unroll
  for (int t = 0; t < 2; ++t) {
    int nt = w * 2 + t;                                // d-tile 0..15
    const unsigned short* brow = ctxT + ((long)b * D_ + nt * 16 + lm) * S_ + 8 * lh;
    const unsigned short* arow = attnS + lm * S_ + 8 * lh;
    v8f acc = {};
    for (int kk = 0; kk < 32; ++kk) {
      FragBF a, bb;
      a.q[0]  = *(const uint4*)(arow + kk * 32);
      a.q[1]  = *(const uint4*)(arow + kk * 32 + 16);
      bb.q[0] = *(const uint4*)(brow + kk * 32);
      bb.q[1] = *(const uint4*)(brow + kk * 32 + 16);
      acc = wmma_bf16(a, bb, acc);
    }
#pragma unroll
    for (int i = 0; i < 8; ++i) qw[(i + 8 * lh) * D_ + nt * 16 + lm] = f2b(acc[i]);
  }
  __syncthreads();

  // ---- Phase 5: h_tilde = tanh([weighted, ctx_q] @ Wo^T); pool over queries ----
#pragma unroll
  for (int t = 0; t < 2; ++t) {
    int nt = w * 2 + t;                                // e-tile 0..15
    const unsigned short* brow = Wo + (long)(nt * 16 + lm) * D2_ + 8 * lh;
    const unsigned short* wrow = qw + lm * D_ + 8 * lh;
    v8f acc = {};
#pragma unroll
    for (int kk = 0; kk < 16; ++kk) {                  // K = 512
      FragBF a, bb;
      if (kk < 8) {                                    // d in [0,256): weighted (LDS)
        a.q[0] = *(const uint4*)(wrow + kk * 32);
        a.q[1] = *(const uint4*)(wrow + kk * 32 + 16);
      } else {                                         // d in [256,512): ctx_q (LDS)
        int k2 = kk - 8;
        a.q[0] = *(const uint4*)(arow0 + k2 * 32);
        a.q[1] = *(const uint4*)(arow0 + k2 * 32 + 16);
      }
      bb.q[0] = *(const uint4*)(brow + kk * 32);
      bb.q[1] = *(const uint4*)(brow + kk * 32 + 16);
      acc = wmma_bf16(a, bb, acc);
    }
    float psum = 0.f;
#pragma unroll
    for (int i = 0; i < 8; ++i) psum += tanhf(acc[i]);
    psum += __shfl_xor(psum, 16, 32);                  // combine m-halves (wave32)
    if (lh == 0) atomicAdd(&pooled[b * D_ + nt * 16 + lm], psum);
  }
}

// ---------------------------------------------------------------------------
// K6: o = tanh(mean_pool @ W_out^T + b_out); BatchNorm over batch (training stats)
// ---------------------------------------------------------------------------
__global__ void final_kernel(const float* __restrict__ pooled,
                             const float* __restrict__ W_out, const float* __restrict__ b_out,
                             const float* __restrict__ gamma, const float* __restrict__ beta,
                             float* __restrict__ out)
{
  int e = threadIdx.x;                                 // 0..127
  float o[B_];
  for (int b = 0; b < B_; ++b) {
    float acc = b_out[e];
    for (int d = 0; d < D_; ++d)
      acc += (pooled[b * D_ + d] * (1.f / (float)S_)) * W_out[e * D_ + d];
    o[b] = tanhf(acc);
  }
  float mu = 0.f;
  for (int b = 0; b < B_; ++b) mu += o[b];
  mu *= (1.f / B_);
  float var = 0.f;
  for (int b = 0; b < B_; ++b) { float d = o[b] - mu; var += d * d; }
  var *= (1.f / B_);
  float inv = rsqrtf(var + 1e-5f);
  for (int b = 0; b < B_; ++b)
    out[b * OUT_ + e] = gamma[e] * (o[b] - mu) * inv + beta[e];
}

// ---------------------------------------------------------------------------
extern "C" void kernel_launch(void* const* d_in, const int* in_sizes, int n_in,
                              void* d_out, int out_size, void* d_ws, size_t ws_size,
                              hipStream_t stream) {
  const int*           ids    = (const int*)d_in[0];
  const unsigned char* mask   = (const unsigned char*)d_in[1];
  const float* table  = (const float*)d_in[2];
  const float* w_ih_f = (const float*)d_in[3];
  const float* w_hh_f = (const float*)d_in[4];
  const float* b_ih_f = (const float*)d_in[5];
  const float* b_hh_f = (const float*)d_in[6];
  const float* w_ih_b = (const float*)d_in[7];
  const float* w_hh_b = (const float*)d_in[8];
  const float* b_ih_b = (const float*)d_in[9];
  const float* b_hh_b = (const float*)d_in[10];
  const float* Wq     = (const float*)d_in[11];
  const float* Wo     = (const float*)d_in[12];
  const float* W_out  = (const float*)d_in[13];
  const float* b_out  = (const float*)d_in[14];
  const float* gamma  = (const float*)d_in[15];
  const float* beta   = (const float*)d_in[16];
  float* out = (float*)d_out;

  size_t off = 0;
  auto alloc = [&](size_t bytes) -> void* {
    off = (off + 255) & ~(size_t)255;
    void* p = (char*)d_ws + off;
    off += bytes;
    return p;
  };
  unsigned short* embeds  = (unsigned short*)alloc((size_t)B_ * S_ * KP_ * 2);
  unsigned short* p_wih_f = (unsigned short*)alloc((size_t)G4_ * KP_ * 2);
  unsigned short* p_wih_b = (unsigned short*)alloc((size_t)G4_ * KP_ * 2);
  unsigned short* p_whh_f = (unsigned short*)alloc((size_t)G4_ * HID_ * 2);
  unsigned short* p_whh_b = (unsigned short*)alloc((size_t)G4_ * HID_ * 2);
  unsigned short* p_wq    = (unsigned short*)alloc((size_t)D_ * D_ * 2);
  unsigned short* p_wo    = (unsigned short*)alloc((size_t)D_ * D2_ * 2);
  unsigned short* xg_f    = (unsigned short*)alloc((size_t)S_ * B_ * G4_ * 2);
  unsigned short* xg_b    = (unsigned short*)alloc((size_t)S_ * B_ * G4_ * 2);
  unsigned short* ctx     = (unsigned short*)alloc((size_t)B_ * S_ * D_ * 2);
  unsigned short* ctxT    = (unsigned short*)alloc((size_t)B_ * D_ * S_ * 2);
  float* pooled = (float*)alloc((size_t)B_ * D_ * 4);

  pack_weights<<<256, 256, 0, stream>>>(w_ih_f, w_hh_f, w_ih_b, w_hh_b, Wq, Wo,
                                        p_wih_f, p_whh_f, p_wih_b, p_whh_b,
                                        p_wq, p_wo, pooled);
  embed_gather<<<B_ * S_, 128, 0, stream>>>(ids, table, embeds);

  dim3 g2(4, S_, 2);
  xg_gemm<<<g2, 256, 0, stream>>>(embeds, p_wih_f, p_wih_b,
                                  b_ih_f, b_hh_f, b_ih_b, b_hh_b, xg_f, xg_b);

  lstm_scan<<<2, 1024, 2 * B_ * G4_ * sizeof(unsigned short), stream>>>(
      xg_f, xg_b, p_whh_f, p_whh_b, ctx);

  dim3 g4(S_ / 32, D_ / 32, B_);
  ctx_transpose<<<g4, dim3(32, 8), 0, stream>>>(ctx, ctxT);

  dim3 g5(S_ / 16, B_);
  attn_kernel<<<g5, 256, 0, stream>>>(ctx, ctxT, p_wq, p_wo, mask, pooled);

  final_kernel<<<1, OUT_, 0, stream>>>(pooled, W_out, b_out, gamma, beta, out);
}
